// ESSM_38603166057068
// MI455X (gfx1250) — matrix operand order repaired and checked by reading, not compiled, not run on hardware
//
#include <hip/hip_runtime.h>
#include <hip/hip_bf16.h>

// ---------------- problem constants ----------------
#define DIMC   64
#define DI     128
#define DSTATE 16
#define DTRANK 4
#define Bb     4
#define Hh     128
#define Ww     128
#define Ll     (Hh*Ww)        // 16384
#define NPIX   (Bb*Ll)        // 65536
#define XDBLP  48             // 36 padded to 3 N-tiles of 16
#define CHUNK  128            // scan chunk length
#define NCH    (Ll/CHUNK)     // 128 chunks per batch image

typedef __attribute__((ext_vector_type(16))) _Float16 v16h;
typedef __attribute__((ext_vector_type(8)))  float    v8f;

// ======================================================================
// Cooperative global -> LDS staging. Uses the gfx1250 async-to-LDS path
// (ASYNCcnt) when the builtin is available; falls back to load+ds_store.
// n must be a multiple of 4 floats; both pointers 16B-aligned.
// ======================================================================
__device__ __forceinline__ void stage_to_lds(float* __restrict__ s,
                                             const float* __restrict__ g, int n) {
#if __has_builtin(__builtin_amdgcn_global_load_async_to_lds_b128)
    typedef __attribute__((__vector_size__(4 * sizeof(int)))) int v4i_;
    typedef __attribute__((address_space(1))) v4i_* gv4i_p;
    typedef __attribute__((address_space(3))) v4i_* lv4i_p;
    for (int i = threadIdx.x * 4; i < n; i += blockDim.x * 4) {
        __builtin_amdgcn_global_load_async_to_lds_b128(
            (gv4i_p)(g + i), (lv4i_p)(s + i), 0, 0);
    }
#if __has_builtin(__builtin_amdgcn_s_wait_asynccnt)
    __builtin_amdgcn_s_wait_asynccnt(0);
#else
    asm volatile("s_wait_asynccnt 0x0" ::: "memory");
#endif
#else
    for (int i = threadIdx.x * 4; i < n; i += blockDim.x * 4) {
        float4 v = *(const float4*)(g + i);
        *(float4*)(s + i) = v;
    }
#endif
    __syncthreads();
}

// ======================================================================
// Weight packing into B fragments for v_wmma_f32_16x16x32_f16 (contiguous k, same order as A):
//   frag f = nt*KT + kt ; per lane 16 contiguous halves
//   lane<16 -> N=nt*16+lane,    K = kt*32 + j      (j=0..15)
//   lane>=16-> N=nt*16+lane-16, K = kt*32 + 16 + j
// ======================================================================
__global__ void pack_b_kernel(const float* __restrict__ W, _Float16* __restrict__ dst,
                              int N, int K, int NT, int KT) {
    int t = blockIdx.x * blockDim.x + threadIdx.x;
    int total = NT * KT * 32;
    if (t >= total) return;
    int lane = t & 31;
    int kt   = (t >> 5) % KT;
    int nt   = (t >> 5) / KT;
    int col  = nt * 16 + (lane & 15);
    int kb   = kt * 32 + ((lane >> 4) << 4);
    _Float16* p = dst + (size_t)t * 16;
#pragma unroll
    for (int j = 0; j < 16; ++j) {
        int k = kb + j;
        float v = (col < N) ? W[(size_t)col * K + k] : 0.0f;  // B[k][n] = W[n][k]
        p[j] = (_Float16)v;
    }
}

// ======================================================================
// LayerNorm over channel dim (64), fp32 in -> fp32 out
// ======================================================================
__global__ void ln_kernel(const float* __restrict__ x, const float* __restrict__ g,
                          const float* __restrict__ b, float* __restrict__ out) {
    int p = blockIdx.x * blockDim.x + threadIdx.x;
    if (p >= NPIX) return;
    const float* row = x + (size_t)p * DIMC;
    float mu = 0.f;
#pragma unroll 8
    for (int c = 0; c < DIMC; ++c) mu += row[c];
    mu *= (1.0f / DIMC);
    float var = 0.f;
#pragma unroll 8
    for (int c = 0; c < DIMC; ++c) { float d = row[c] - mu; var += d * d; }
    var *= (1.0f / DIMC);
    float rs = rsqrtf(var + 1e-5f);
    float* o = out + (size_t)p * DIMC;
#pragma unroll 8
    for (int c = 0; c < DIMC; ++c) o[c] = (row[c] - mu) * rs * g[c] + b[c];
}

// ======================================================================
// WMMA GEMM: C(M x 16*NT) = A(M x 32*KT, fp32 row-major) * packedB (+ Res)
// One wave per 16-row M tile; A fragments in VGPRs across all N tiles.
// LDA/LDC/NT are compile-time so every load/store folds into an
// immediate-offset vmem op. grid = M/128 blocks of 256 threads (8 waves).
// ======================================================================
template<int KT, int NT, int LDA, int LDC, bool ADD_RES>
__global__ void gemm_wmma_kernel(const float* __restrict__ A, const _Float16* __restrict__ Bp,
                                 float* __restrict__ C, const float* __restrict__ Res) {
    const int lane = threadIdx.x & 31;
    const int wv   = threadIdx.x >> 5;
    const int m0   = (blockIdx.x * 8 + wv) * 16;
    const int row  = m0 + (lane & 15);
    const int kh   = (lane >> 4) << 4;       // 0 or 16 (A half-K offset)

    v16h afrag[KT];
    const float* pa0 = A + (size_t)row * LDA + kh;
#pragma unroll
    for (int kt = 0; kt < KT; ++kt) {
        v16h a;
#pragma unroll
        for (int j = 0; j < 8; ++j) a[j]     = (_Float16)pa0[kt * 32 + j];
#pragma unroll
        for (int j = 0; j < 8; ++j) a[8 + j] = (_Float16)pa0[kt * 32 + 8 + j];
        afrag[kt] = a;
    }
    // speculative prefetch of the next block's A rows (global_prefetch path)
    __builtin_prefetch(A + ((size_t)row + 128) * LDA, 0, 1);

    const int n    = (lane & 15);
    const int mrb  = m0 + ((lane >> 4) << 3);
    const _Float16* bp = Bp + (size_t)lane * 16;
#pragma unroll
    for (int nt = 0; nt < NT; ++nt) {
        v8f acc = {};
#pragma unroll
        for (int kt = 0; kt < KT; ++kt) {
            v16h bf = *(const v16h*)(bp + (size_t)(nt * KT + kt) * 32 * 16);
            acc = __builtin_amdgcn_wmma_f32_16x16x32_f16(
                false, afrag[kt], false, bf, (short)0, acc, false, false);
        }
        const int ncol = nt * 16 + n;
        float* cp = C + (size_t)mrb * LDC + ncol;
        const float* rp = Res + (size_t)mrb * LDC + ncol;
#pragma unroll
        for (int r = 0; r < 8; ++r) {
            float v = acc[r];
            if (ADD_RES) v += rp[r * LDC];
            cp[r * LDC] = v;
        }
    }
}

// ======================================================================
// Depthwise 3x3 conv (SAME), optional exact GELU, optional residual add
// ======================================================================
template<bool GELU_, bool ADD_RES>
__global__ void dw3_kernel(const float* __restrict__ in, const float* __restrict__ wt,
                           float* __restrict__ out, const float* __restrict__ Res) {
    int idx = blockIdx.x * blockDim.x + threadIdx.x;   // over NPIX*DIMC
    if (idx >= NPIX * DIMC) return;
    int c = idx & (DIMC - 1);
    int p = idx >> 6;
    int w = p % Ww;
    int h = (p / Ww) % Hh;
    int b = p / (Ww * Hh);
    float acc = 0.f;
#pragma unroll
    for (int i = 0; i < 3; ++i) {
        int hh = h + i - 1;
        if (hh < 0 || hh >= Hh) continue;
#pragma unroll
        for (int j = 0; j < 3; ++j) {
            int ww = w + j - 1;
            if (ww < 0 || ww >= Ww) continue;
            acc += in[((((size_t)b * Hh + hh) * Ww + ww) << 6) + c] * wt[(c * 3 + i) * 3 + j];
        }
    }
    if (GELU_) acc = 0.5f * acc * (1.f + erff(acc * 0.70710678118f));
    if (ADD_RES) acc += Res[idx];
    out[idx] = acc;
}

// ======================================================================
// Causal depthwise conv1d (k=4) over L + SiLU.  xs = xz[..., 0:128]
// ======================================================================
__global__ void cconv_silu_kernel(const float* __restrict__ xz, const float* __restrict__ cw,
                                  const float* __restrict__ cb, float* __restrict__ u) {
    int idx = blockIdx.x * blockDim.x + threadIdx.x;   // over NPIX*DI
    if (idx >= NPIX * DI) return;
    int d = idx & (DI - 1);
    int m = idx >> 7;
    int l = m % Ll;
    int b = m / Ll;
    float acc = cb[d];
#pragma unroll
    for (int j = 0; j < 4; ++j) {
        int ls = l - 3 + j;
        if (ls < 0) continue;
        acc += xz[((size_t)(b * Ll + ls)) * 256 + d] * cw[d * 4 + j];
    }
    u[idx] = acc / (1.f + __expf(-acc));
}

// ======================================================================
// dt = softplus(dt_r @ W_dt^T + b_dt)   (K=4, VALU)
// ======================================================================
__global__ void dt_kernel(const float* __restrict__ xdbl, const float* __restrict__ Wdt,
                          const float* __restrict__ bdt, float* __restrict__ dt) {
    int idx = blockIdx.x * blockDim.x + threadIdx.x;   // over NPIX*DI
    if (idx >= NPIX * DI) return;
    int d = idx & (DI - 1);
    int m = idx >> 7;
    float acc = bdt[d];
#pragma unroll
    for (int r = 0; r < DTRANK; ++r)
        acc += xdbl[(size_t)m * XDBLP + r] * Wdt[d * DTRANK + r];
    dt[idx] = (acc > 30.f) ? acc : log1pf(__expf(acc));
}

// ======================================================================
// Chunked selective scan.
//  - one block (128 threads = 4 waves) per (b,chunk); thread = channel d
//  - per-chunk x_dbl slab (128x48 = 24 KB) staged in LDS (async-to-LDS)
//  - 16 states per thread in registers; dt/u reads coalesced across d
// pass1: P = prod(dA), h_end with h0=0
// pass2: per (b,d,s) scan over chunks -> h0 per chunk
// pass3: replay with true h0, emit y; fuse +u*D and *silu(z)
// ======================================================================
__global__ void scan_pass1_kernel(const float* __restrict__ dt, const float* __restrict__ u,
                                  const float* __restrict__ xdbl, const float* __restrict__ A_log,
                                  float* __restrict__ P, float* __restrict__ Hend) {
    __shared__ __attribute__((aligned(16))) float sx[CHUNK * XDBLP];
    const int blk  = blockIdx.x;              // b*NCH + ck
    const int base = blk * CHUNK;             // global step index (b*Ll + ck*CHUNK)
    stage_to_lds(sx, xdbl + (size_t)base * XDBLP, CHUNK * XDBLP);

    const int d = threadIdx.x;                // 0..127
    float a[DSTATE], h[DSTATE], Pa[DSTATE];
#pragma unroll
    for (int s = 0; s < DSTATE; ++s) {
        a[s]  = -__expf(A_log[d * DSTATE + s]);
        h[s]  = 0.f;
        Pa[s] = 1.f;
    }
    for (int i = 0; i < CHUNK; ++i) {
        const int m   = base + i;
        const float dtv = dt[(size_t)m * DI + d];
        const float uv  = u[(size_t)m * DI + d];
        const float du  = dtv * uv;
        const float* xr = sx + i * XDBLP;
#pragma unroll
        for (int s = 0; s < DSTATE; ++s) {
            float dA = __expf(dtv * a[s]);
            Pa[s] *= dA;
            h[s] = dA * h[s] + du * xr[DTRANK + s];
        }
    }
    const size_t o = ((size_t)blk * DI + d) * DSTATE;
#pragma unroll
    for (int s = 0; s < DSTATE; ++s) { P[o + s] = Pa[s]; Hend[o + s] = h[s]; }
}

__global__ void scan_pass2_kernel(const float* __restrict__ P, const float* __restrict__ Hend,
                                  float* __restrict__ H0) {
    int t = blockIdx.x * blockDim.x + threadIdx.x;     // Bb*DI*DSTATE = 8192
    if (t >= Bb * DI * DSTATE) return;
    int s = t & 15;
    int d = (t >> 4) & (DI - 1);
    int b = t >> 11;
    float h = 0.f;
    for (int ck = 0; ck < NCH; ++ck) {
        size_t idx = ((((size_t)(b * NCH + ck)) * DI + d) << 4) + s;
        H0[idx] = h;
        h = P[idx] * h + Hend[idx];
    }
}

__global__ void scan_pass3_kernel(const float* __restrict__ dt, const float* __restrict__ u,
                                  const float* __restrict__ xdbl, const float* __restrict__ xz,
                                  const float* __restrict__ A_log, const float* __restrict__ Dp,
                                  const float* __restrict__ H0, float* __restrict__ y) {
    __shared__ __attribute__((aligned(16))) float sx[CHUNK * XDBLP];
    const int blk  = blockIdx.x;              // b*NCH + ck
    const int base = blk * CHUNK;
    stage_to_lds(sx, xdbl + (size_t)base * XDBLP, CHUNK * XDBLP);

    const int d = threadIdx.x;
    float a[DSTATE], h[DSTATE];
#pragma unroll
    for (int s = 0; s < DSTATE; ++s) {
        a[s] = -__expf(A_log[d * DSTATE + s]);
        h[s] = H0[(((size_t)blk * DI + d) << 4) + s];
    }
    const float Dd = Dp[d];
    for (int i = 0; i < CHUNK; ++i) {
        const int m   = base + i;
        const float dtv = dt[(size_t)m * DI + d];
        const float uv  = u[(size_t)m * DI + d];
        const float du  = dtv * uv;
        const float* xr = sx + i * XDBLP;
        float yv = 0.f;
#pragma unroll
        for (int s = 0; s < DSTATE; ++s) {
            float dA = __expf(dtv * a[s]);
            h[s] = dA * h[s] + du * xr[DTRANK + s];
            yv += h[s] * xr[DTRANK + DSTATE + s];
        }
        yv += uv * Dd;
        const float zv = xz[(size_t)m * 256 + DI + d];
        yv *= zv / (1.f + __expf(-zv));
        y[(size_t)m * DI + d] = yv;
    }
}

// ======================================================================
// Host-side orchestration
// ======================================================================
extern "C" void kernel_launch(void* const* d_in, const int* in_sizes, int n_in,
                              void* d_out, int out_size, void* d_ws, size_t ws_size,
                              hipStream_t stream) {
    const float* x        = (const float*)d_in[0];
    const float* ln_g     = (const float*)d_in[1];
    const float* ln_b     = (const float*)d_in[2];
    const float* w_in_pw1 = (const float*)d_in[3];
    const float* w_in_dw  = (const float*)d_in[4];
    const float* w_in_pw2 = (const float*)d_in[5];
    const float* w_out_dw1= (const float*)d_in[6];
    const float* w_out_dw2= (const float*)d_in[7];
    const float* W_inproj = (const float*)d_in[8];
    const float* conv_w   = (const float*)d_in[9];
    const float* conv_b   = (const float*)d_in[10];
    const float* W_xproj  = (const float*)d_in[11];
    const float* W_dt     = (const float*)d_in[12];
    const float* b_dt     = (const float*)d_in[13];
    const float* A_log    = (const float*)d_in[14];
    const float* Dp       = (const float*)d_in[15];
    const float* W_out    = (const float*)d_in[16];
    float* out = (float*)d_out;

    char* ws = (char*)d_ws;
    size_t off = 0;
    auto allocf = [&](size_t nf) -> float* {
        float* p = (float*)(ws + off);
        off += ((nf * sizeof(float) + 255) / 256) * 256;
        return p;
    };
    auto alloch = [&](size_t nh) -> _Float16* {
        _Float16* p = (_Float16*)(ws + off);
        off += ((nh * sizeof(_Float16) + 255) / 256) * 256;
        return p;
    };

    float* xn   = allocf((size_t)NPIX * DIMC);
    float* t1   = allocf((size_t)NPIX * DIMC);
    float* t2   = allocf((size_t)NPIX * DIMC);
    float* x0   = allocf((size_t)NPIX * DIMC);
    float* xz   = allocf((size_t)NPIX * 256);
    float* u    = allocf((size_t)NPIX * DI);
    float* xdbl = allocf((size_t)NPIX * XDBLP);
    float* dtb  = allocf((size_t)NPIX * DI);
    float* yb   = allocf((size_t)NPIX * DI);
    float* x1   = allocf((size_t)NPIX * DIMC);
    const size_t nscan = (size_t)Bb * NCH * DI * DSTATE;
    float* Ps   = allocf(nscan);
    float* Hend = allocf(nscan);
    float* H0   = allocf(nscan);
    _Float16* wp_pw1 = alloch(4  * 2 * 32 * 16);   // 64x64
    _Float16* wp_pw2 = alloch(4  * 2 * 32 * 16);   // 64x64
    _Float16* wp_inp = alloch(16 * 2 * 32 * 16);   // 256x64
    _Float16* wp_xp  = alloch(3  * 4 * 32 * 16);   // 36(->48)x128
    _Float16* wp_out = alloch(4  * 4 * 32 * 16);   // 64x128
    float* tmp = t1;  // t1 dead after pw2; reuse for final dw intermediate

    // ---- pack weights into WMMA B-fragment layout ----
    pack_b_kernel<<<1, 256, 0, stream>>>(w_in_pw1, wp_pw1, 64, 64, 4, 2);
    pack_b_kernel<<<1, 256, 0, stream>>>(w_in_pw2, wp_pw2, 64, 64, 4, 2);
    pack_b_kernel<<<4, 256, 0, stream>>>(W_inproj, wp_inp, 256, 64, 16, 2);
    pack_b_kernel<<<2, 256, 0, stream>>>(W_xproj,  wp_xp,  36, 128, 3, 4);
    pack_b_kernel<<<2, 256, 0, stream>>>(W_out,    wp_out, 64, 128, 4, 4);

    // ---- stage 1: LN -> pw1 (WMMA) -> dw3+GELU -> pw2 (WMMA) ----
    ln_kernel<<<NPIX / 256, 256, 0, stream>>>(x, ln_g, ln_b, xn);
    gemm_wmma_kernel<2, 4, 64, 64, false><<<NPIX / 128, 256, 0, stream>>>(xn, wp_pw1, t1, nullptr);
    dw3_kernel<true, false><<<(NPIX * DIMC) / 256, 256, 0, stream>>>(t1, w_in_dw, t2, nullptr);
    gemm_wmma_kernel<2, 4, 64, 64, false><<<NPIX / 128, 256, 0, stream>>>(t2, wp_pw2, x0, nullptr);

    // ---- stage 2: in-proj GEMM, causal conv + SiLU, x-proj GEMM, dt ----
    gemm_wmma_kernel<2, 16, 64, 256, false><<<NPIX / 128, 256, 0, stream>>>(x0, wp_inp, xz, nullptr);
    cconv_silu_kernel<<<(NPIX * DI) / 256, 256, 0, stream>>>(xz, conv_w, conv_b, u);
    gemm_wmma_kernel<4, 3, 128, XDBLP, false><<<NPIX / 128, 256, 0, stream>>>(u, wp_xp, xdbl, nullptr);
    dt_kernel<<<(NPIX * DI) / 256, 256, 0, stream>>>(xdbl, W_dt, b_dt, dtb);

    // ---- stage 3: chunked selective scan (3 passes), gating fused in pass3 ----
    scan_pass1_kernel<<<Bb * NCH, DI, 0, stream>>>(dtb, u, xdbl, A_log, Ps, Hend);
    scan_pass2_kernel<<<(Bb * DI * DSTATE) / 256, 256, 0, stream>>>(Ps, Hend, H0);
    scan_pass3_kernel<<<Bb * NCH, DI, 0, stream>>>(dtb, u, xdbl, xz, A_log, Dp, H0, yb);

    // ---- stage 4: out-proj GEMM + residual x0 -> x1 ----
    gemm_wmma_kernel<4, 4, 128, 64, true><<<NPIX / 128, 256, 0, stream>>>(yb, wp_out, x1, x0);

    // ---- stage 5: dw3 -> GELU -> dw3 -> + x1 ----
    dw3_kernel<true, false><<<(NPIX * DIMC) / 256, 256, 0, stream>>>(x1, w_out_dw1, tmp, nullptr);
    dw3_kernel<false, true><<<(NPIX * DIMC) / 256, 256, 0, stream>>>(tmp, w_out_dw2, out, x1);
}
